// CPN_41858751267015
// MI455X (gfx1250) — matrix operand order repaired and checked by reading, not compiled, not run on hardware
//
#include <hip/hip_runtime.h>
#include <hip/hip_bf16.h>
#include <math.h>

// Problem sizes (fixed by the reference)
#define B_N 16384
#define D_N 256
#define K_N 8192
#define O_N 128
#define KC_TILES (K_N / 16)

typedef __attribute__((ext_vector_type(16))) __bf16        v16bf;
typedef __attribute__((ext_vector_type(8)))  float         v8f;
typedef __attribute__((ext_vector_type(4)))  unsigned int  v4u;
typedef __attribute__((__vector_size__(16))) int           v4i;   // builtin's b128 type

union frag_u { v16bf v; v4u u[2]; };

// CDNA5 async global->LDS path (ASYNCcnt), guarded so a missing builtin
// falls back to the proven direct-global path instead of failing the build.
#if defined(__has_builtin)
#if __has_builtin(__builtin_amdgcn_global_load_async_to_lds_b128) && \
    __has_builtin(__builtin_amdgcn_s_wait_asynccnt)
#define USE_ASYNC_LDS 1
#endif
#endif
#ifndef USE_ASYNC_LDS
#define USE_ASYNC_LDS 0
#endif

#if USE_ASYNC_LDS
// exact pointer types the builtin expects (from the round-2 diagnostic):
//   arg0: v4i addrspace(1)*   arg1: v4i addrspace(3)*   args 2,3: imm ints
#define ASYNC_GP(p) ((__attribute__((address_space(1))) v4i*)(p))
#define ASYNC_LP(p) ((__attribute__((address_space(3))) v4i*)(p))
#endif

__device__ __forceinline__ unsigned short f32_to_bf16_rne(float f) {
  unsigned int u = __float_as_uint(f);
  u += 0x7fffu + ((u >> 16) & 1u);            // round-to-nearest-even
  return (unsigned short)(u >> 16);
}
__device__ __forceinline__ float bf16_to_f32(unsigned short h) {
  return __uint_as_float(((unsigned int)h) << 16);
}

// ---------------------------------------------------------------------------
// Stage 1: L2-normalize x rows, split into bf16 hi/lo planes.
// ---------------------------------------------------------------------------
__global__ void cpn_prep_x(const float* __restrict__ x,
                           unsigned short* __restrict__ xhi,
                           unsigned short* __restrict__ xlo) {
  __shared__ float red[256];
  const int b = blockIdx.x, t = threadIdx.x;
  const float v = x[(size_t)b * D_N + t];
  red[t] = v * v;
  __syncthreads();
  for (int s = 128; s > 0; s >>= 1) {
    if (t < s) red[t] += red[t + s];
    __syncthreads();
  }
  const float inv = 1.0f / fmaxf(sqrtf(red[0]), 1e-12f);
  const float xn = v * inv;
  const unsigned short hi = f32_to_bf16_rne(xn);
  const float lo = xn - bf16_to_f32(hi);
  xhi[(size_t)b * D_N + t] = hi;
  xlo[(size_t)b * D_N + t] = f32_to_bf16_rne(lo);
}

// ---------------------------------------------------------------------------
// Stage 2: split kw into bf16 hi/lo planes + per-row sum of squares.
// ---------------------------------------------------------------------------
__global__ void cpn_prep_w(const float* __restrict__ kw,
                           unsigned short* __restrict__ whi,
                           unsigned short* __restrict__ wlo,
                           float* __restrict__ wnorm) {
  __shared__ float red[256];
  const int k = blockIdx.x, t = threadIdx.x;
  const float v = kw[(size_t)k * D_N + t];
  red[t] = v * v;
  __syncthreads();
  for (int s = 128; s > 0; s >>= 1) {
    if (t < s) red[t] += red[t + s];
    __syncthreads();
  }
  if (t == 0) wnorm[k] = red[0];
  const unsigned short hi = f32_to_bf16_rne(v);
  const float lo = v - bf16_to_f32(hi);
  whi[(size_t)k * D_N + t] = hi;
  wlo[(size_t)k * D_N + t] = f32_to_bf16_rne(lo);
}

// ---------------------------------------------------------------------------
// Stage 3: transpose grossberg_w [O,K] -> gwT [K,O].
// ---------------------------------------------------------------------------
__global__ void cpn_transpose_gw(const float* __restrict__ gw,
                                 float* __restrict__ gwT) {
  __shared__ float tile[32][33];
  const int k0 = blockIdx.x * 32, o0 = blockIdx.y * 32;
  const int tx = threadIdx.x, ty = threadIdx.y;
  for (int i = ty; i < 32; i += 8)
    tile[i][tx] = gw[(size_t)(o0 + i) * K_N + (k0 + tx)];
  __syncthreads();
  for (int i = ty; i < 32; i += 8)
    gwT[(size_t)(k0 + i) * O_N + (o0 + tx)] = tile[tx][i];
}

// ---------------------------------------------------------------------------
// Stage 4 core: score_j = wnorm[j] - 2*(xn_b . kw_j) via split-bf16 WMMA,
// running argmin over K. 4 waves/WG, 16 batch rows per wave, A fragments
// register-resident; B tiles staged once per WG in LDS via the CDNA5 async
// global->LDS engine (double-buffered), falling back to direct global loads.
// ---------------------------------------------------------------------------
#if USE_ASYNC_LDS
#define ROW_PAD 264   // 256 + 8 bf16: 528B row stride = 132 dwords (=4 mod 64
                      // banks) -> 32-lane b128 reads hit each bank exactly 2x

__device__ __forceinline__ void issue_tile_load(
    const unsigned short* __restrict__ whi,
    const unsigned short* __restrict__ wlo,
    unsigned short (*buf)[16][ROW_PAD],   // buf[plane][row][col]
    int kc, int tid) {
  const size_t gbase = (size_t)kc * 16 * D_N;
#pragma unroll
  for (int i = 0; i < 4; ++i) {
    const int chunk = tid + i * 128;      // 512 x 16B chunks per plane
    const int row   = chunk >> 5;
    const int col   = (chunk & 31) * 8;   // element offset
    const unsigned short* ghi = whi + gbase + (size_t)row * D_N + col;
    const unsigned short* glo = wlo + gbase + (size_t)row * D_N + col;
    __builtin_amdgcn_global_load_async_to_lds_b128(
        ASYNC_GP(ghi), ASYNC_LP(&buf[0][row][col]), 0, 0);
    __builtin_amdgcn_global_load_async_to_lds_b128(
        ASYNC_GP(glo), ASYNC_LP(&buf[1][row][col]), 0, 0);
  }
}
#endif

__global__ void __launch_bounds__(128, 1)
cpn_gemm_argmin(const unsigned short* __restrict__ xhi,
                const unsigned short* __restrict__ xlo,
                const unsigned short* __restrict__ whi,
                const unsigned short* __restrict__ wlo,
                const float* __restrict__ wnorm,
                int* __restrict__ winners) {
  const int tid  = threadIdx.x;
  const int lane = tid & 31;
  const int wave = tid >> 5;
  const int hsel = lane >> 4;   // wave32 half
  const int l15  = lane & 15;
  const int rowBase = blockIdx.x * 64 + wave * 16;

  // ---- A fragments: 16 rows x 256 D, hi+lo, register-resident (128 VGPRs)
  // A-layout (16x32 bf16): lane L = row (L&15); half0 k=0..15,
  // half1 k=16..31 -> one contiguous 32B run per lane per 32-d step.
  frag_u ah[8], al[8];
  {
    const size_t rb = (size_t)(rowBase + l15) * D_N;
#pragma unroll
    for (int s = 0; s < 8; ++s) {
      const int d0 = s * 32 + hsel * 16;
      ah[s].u[0] = *reinterpret_cast<const v4u*>(xhi + rb + d0);
      ah[s].u[1] = *reinterpret_cast<const v4u*>(xhi + rb + d0 + 8);
      al[s].u[0] = *reinterpret_cast<const v4u*>(xlo + rb + d0);
      al[s].u[1] = *reinterpret_cast<const v4u*>(xlo + rb + d0 + 8);
    }
  }

  float minv[8];
  int   mini[8];
#pragma unroll
  for (int r = 0; r < 8; ++r) { minv[r] = 3.0e38f; mini[r] = 0; }

#if USE_ASYNC_LDS
  // double-buffered B tile: [buf][plane(hi/lo)][row 16][ROW_PAD]
  __shared__ unsigned short bt[2][2][16][ROW_PAD];
  issue_tile_load(whi, wlo, bt[0], 0, tid);
  __builtin_amdgcn_s_wait_asynccnt(0);
  __syncthreads();
#else
  const size_t bstride = (size_t)16 * D_N;
  const size_t bbase   = (size_t)l15 * D_N + hsel * 16;
#endif

  for (int kc = 0; kc < KC_TILES; ++kc) {
    v8f c0 = {0.f,0.f,0.f,0.f,0.f,0.f,0.f,0.f};   // hi*hi
    v8f c1 = {0.f,0.f,0.f,0.f,0.f,0.f,0.f,0.f};   // hi*lo
    v8f c2 = {0.f,0.f,0.f,0.f,0.f,0.f,0.f,0.f};   // lo*hi

#if USE_ASYNC_LDS
    const int cur = kc & 1;
    if (kc + 1 < KC_TILES)
      issue_tile_load(whi, wlo, bt[cur ^ 1], kc + 1, tid);
    // B-layout used here, same k order as A (32x16 bf16): lane L = column (L&15); half0 k=0..15,
    // half1 k=16..31 -> one contiguous 32B run per lane per 32-d step.
    const unsigned short* bhrow = &bt[cur][0][l15][hsel * 16];
    const unsigned short* blrow = &bt[cur][1][l15][hsel * 16];
#pragma unroll
    for (int s = 0; s < 8; ++s) {
      const int d0 = s * 32;
      frag_u bh, bl;
      bh.u[0] = *reinterpret_cast<const v4u*>(bhrow + d0);
      bh.u[1] = *reinterpret_cast<const v4u*>(bhrow + d0 + 8);
      bl.u[0] = *reinterpret_cast<const v4u*>(blrow + d0);
      bl.u[1] = *reinterpret_cast<const v4u*>(blrow + d0 + 8);
      c0 = __builtin_amdgcn_wmma_f32_16x16x32_bf16(false, ah[s].v, false, bh.v,
                                                   (short)0, c0, false, false);
      c1 = __builtin_amdgcn_wmma_f32_16x16x32_bf16(false, ah[s].v, false, bl.v,
                                                   (short)0, c1, false, false);
      c2 = __builtin_amdgcn_wmma_f32_16x16x32_bf16(false, al[s].v, false, bh.v,
                                                   (short)0, c2, false, false);
    }
#else
    const size_t boff = (size_t)kc * bstride + bbase;
    if (kc + 1 < KC_TILES) {                 // WGP-scope prefetch of next tile
      __builtin_prefetch(whi + boff + bstride, 0, 3);
      __builtin_prefetch(wlo + boff + bstride, 0, 3);
    }
#pragma unroll
    for (int s = 0; s < 8; ++s) {
      const int d0 = s * 32;
      frag_u bh, bl;
      bh.u[0] = *reinterpret_cast<const v4u*>(whi + boff + d0);
      bh.u[1] = *reinterpret_cast<const v4u*>(whi + boff + d0 + 8);
      bl.u[0] = *reinterpret_cast<const v4u*>(wlo + boff + d0);
      bl.u[1] = *reinterpret_cast<const v4u*>(wlo + boff + d0 + 8);
      c0 = __builtin_amdgcn_wmma_f32_16x16x32_bf16(false, ah[s].v, false, bh.v,
                                                   (short)0, c0, false, false);
      c1 = __builtin_amdgcn_wmma_f32_16x16x32_bf16(false, ah[s].v, false, bl.v,
                                                   (short)0, c1, false, false);
      c2 = __builtin_amdgcn_wmma_f32_16x16x32_bf16(false, al[s].v, false, bh.v,
                                                   (short)0, c2, false, false);
    }
#endif

    const float wn  = wnorm[kc * 16 + l15];
    const int   idx = kc * 16 + l15;
#pragma unroll
    for (int r = 0; r < 8; ++r) {
      const float s3 = c0[r] + c1[r] + c2[r];
      const float sc = fmaf(-2.0f, s3, wn);
      if (sc < minv[r]) { minv[r] = sc; mini[r] = idx; }  // earliest kc wins ties
    }

#if USE_ASYNC_LDS
    __builtin_amdgcn_s_wait_asynccnt(0);   // our copies for kc+1 landed
    __syncthreads();                       // all waves' copies + DS reads done
#endif
  }

  // argmin across the 16 N-lanes of each half (value, then lowest index)
#pragma unroll
  for (int m = 1; m <= 8; m <<= 1) {
#pragma unroll
    for (int r = 0; r < 8; ++r) {
      const float ov = __shfl_xor(minv[r], m, 32);
      const int   oi = __shfl_xor(mini[r], m, 32);
      if (ov < minv[r] || (ov == minv[r] && oi < mini[r])) {
        minv[r] = ov; mini[r] = oi;
      }
    }
  }
  if (l15 == 0) {
#pragma unroll
    for (int r = 0; r < 8; ++r)
      winners[rowBase + hsel * 8 + r] = mini[r];
  }
}

// ---------------------------------------------------------------------------
// Stage 5: out[b,o] = sigmoid(gwT[win_b, o] + gb[o])
// ---------------------------------------------------------------------------
__global__ void cpn_output(const float* __restrict__ gwT,
                           const float* __restrict__ gb,
                           const int* __restrict__ winners,
                           float* __restrict__ out) {
  const int b = blockIdx.x;
  const int o = threadIdx.x;
  const int win = winners[b];
  const float v = gwT[(size_t)win * O_N + o] + gb[o];
  out[(size_t)b * O_N + o] = 1.0f / (1.0f + expf(-v));
}

// ---------------------------------------------------------------------------
extern "C" void kernel_launch(void* const* d_in, const int* in_sizes, int n_in,
                              void* d_out, int out_size, void* d_ws, size_t ws_size,
                              hipStream_t stream) {
  const float* x  = (const float*)d_in[0];   // [B, D]
  const float* kw = (const float*)d_in[1];   // [K, D]
  const float* gw = (const float*)d_in[2];   // [O, K]
  const float* gb = (const float*)d_in[3];   // [O]

  // workspace layout (all 16B-aligned; ~28 MB total)
  unsigned short* xhi = (unsigned short*)d_ws;               // B*D bf16
  unsigned short* xlo = xhi + (size_t)B_N * D_N;             // B*D bf16
  unsigned short* whi = xlo + (size_t)B_N * D_N;             // K*D bf16
  unsigned short* wlo = whi + (size_t)K_N * D_N;             // K*D bf16
  float* wnorm = (float*)(wlo + (size_t)K_N * D_N);          // K f32
  float* gwT   = wnorm + K_N;                                // K*O f32

  float* out     = (float*)d_out;                            // [B, O] f32
  int*   winners = (int*)(out + (size_t)B_N * O_N);          // [B] int32

  cpn_prep_x<<<B_N, 256, 0, stream>>>(x, xhi, xlo);
  cpn_prep_w<<<K_N, 256, 0, stream>>>(kw, whi, wlo, wnorm);
  cpn_transpose_gw<<<dim3(K_N / 32, O_N / 32), dim3(32, 8), 0, stream>>>(gw, gwT);
  cpn_gemm_argmin<<<B_N / 64, 128, 0, stream>>>(xhi, xlo, whi, wlo, wnorm, winners);
  cpn_output<<<B_N, O_N, 0, stream>>>(gwT, gb, winners, out);
}
